// SDT_48498770706534
// MI455X (gfx1250) — compile-verified, matrix operand order checked by reading, not run on hardware
//
#include <hip/hip_runtime.h>
#include <hip/hip_bf16.h>

typedef __attribute__((ext_vector_type(16))) __bf16 v16bf;
typedef __attribute__((ext_vector_type(8)))  float  v8f;

#define BATCH   32768
#define DIM     784
#define DP1     785
#define KSTEPS  25          // K padded to 800 = 25 * 32

// workspace layout (bytes)
#define WS_PT    0                    // p transposed [1024][32768] f32 = 134217728
#define WS_WP    134217728            // packed W bf16: 64 ntiles * 25 ksteps * 32 lanes * 16 = 1638400 B
#define WS_BIAS  135856128            // bias [1024] f32
#define WS_SPART 135860224            // per-block leaf sums [64][1024] f32 = 262144 B

__device__ __forceinline__ float sigf(float x) {
  return 1.0f / (1.0f + __expf(-x));
}

// ---------------------------------------------------------------------------
// Kernel 1: pack W_inner (f32, [1023][785]) into bf16 WMMA B-fragment layout.
// Fragment layout per 16x32 (NxK) tile: lane = (half<<4)|n16 holds N = n16,
// elements 0..7 = K base kb0..kb0+7, elements 8..15 = K kb0+16..kb0+23,
// kb0 = half*8. Also extracts bias column (W_inner[:,0]).
// ---------------------------------------------------------------------------
__global__ __launch_bounds__(256)
void sdt_prepack(const float* __restrict__ Wi,
                 __bf16* __restrict__ Wp,
                 float* __restrict__ bias) {
  const int g = blockIdx.x * 256 + threadIdx.x;
  if (g < 64 * KSTEPS * 32) {                  // 51200 fragment-lane groups
    const int lane  = g & 31;
    const int ks    = (g >> 5) % KSTEPS;
    const int ntile = g / (KSTEPS * 32);
    const int n   = ntile * 16 + (lane & 15);
    const int kb0 = (lane >> 4) * 8;
    __bf16* dst = Wp + (size_t)g * 16;
#pragma unroll
    for (int e = 0; e < 16; ++e) {
      const int kg = ks * 32 + kb0 + ((e < 8) ? e : (e + 8));
      float v = 0.0f;
      if (n < 1023 && kg < DIM) v = Wi[(size_t)n * DP1 + 1 + kg];
      dst[e] = (__bf16)v;
    }
  }
  if (g < 1024) bias[g] = (g < 1023) ? Wi[(size_t)g * DP1] : 0.0f;
}

// ---------------------------------------------------------------------------
// Kernel 2: p_T = sigmoid(X @ Wi^T + bias), stored transposed [node][batch].
// Block = 256 threads (8 waves): 4 waves x 2 waves tile a 128(M) x 64(N)
// block. Each wave: 32x32 tile = 4 WMMA accumulators, 25 K-steps of K=32.
// X converted f32 -> bf16 in-register; bias preloaded into C.
// ---------------------------------------------------------------------------
__global__ __launch_bounds__(256)
void sdt_gemm_sigmoid(const float* __restrict__ X,
                      const __bf16* __restrict__ Wp,
                      const float* __restrict__ bias,
                      float* __restrict__ pT) {
  const int lane = threadIdx.x & 31;
  const int wave = threadIdx.x >> 5;
  const int wm = wave & 3;
  const int wn = wave >> 2;
  const int half = lane >> 4;
  const int mrow = lane & 15;
  const int Rbase = blockIdx.y * 128 + wm * 32;
  const int NT = blockIdx.x * 4 + wn * 2;       // base 16-wide N-tile index

  v8f acc[2][2];
#pragma unroll
  for (int mt = 0; mt < 2; ++mt)
#pragma unroll
    for (int nt = 0; nt < 2; ++nt) {
      const float bv = bias[(NT + nt) * 16 + mrow];  // this lane's N column
      acc[mt][nt] = (v8f){bv, bv, bv, bv, bv, bv, bv, bv};
    }

  const float* Xr[2];
  Xr[0] = X + (size_t)(Rbase + mrow) * DIM;
  Xr[1] = X + (size_t)(Rbase + 16 + mrow) * DIM;

  for (int ks = 0; ks < KSTEPS; ++ks) {
    const int KB = ks * 32 + half * 8;
    v16bf a[2];
#pragma unroll
    for (int mt = 0; mt < 2; ++mt) {
      const float* src = Xr[mt] + KB;
      float4 c0 = *(const float4*)(src);
      float4 c1 = *(const float4*)(src + 4);
      float4 c2, c3;
      if (ks < KSTEPS - 1) {          // last step: K in [784,800) is padding
        c2 = *(const float4*)(src + 16);
        c3 = *(const float4*)(src + 20);
      } else {
        c2 = make_float4(0.f, 0.f, 0.f, 0.f);
        c3 = c2;
      }
      v16bf av;
      av[0]  = (__bf16)c0.x; av[1]  = (__bf16)c0.y;
      av[2]  = (__bf16)c0.z; av[3]  = (__bf16)c0.w;
      av[4]  = (__bf16)c1.x; av[5]  = (__bf16)c1.y;
      av[6]  = (__bf16)c1.z; av[7]  = (__bf16)c1.w;
      av[8]  = (__bf16)c2.x; av[9]  = (__bf16)c2.y;
      av[10] = (__bf16)c2.z; av[11] = (__bf16)c2.w;
      av[12] = (__bf16)c3.x; av[13] = (__bf16)c3.y;
      av[14] = (__bf16)c3.z; av[15] = (__bf16)c3.w;
      a[mt] = av;
    }
    v16bf bfrag[2];
#pragma unroll
    for (int nt = 0; nt < 2; ++nt) {
      const __bf16* bp = Wp + ((size_t)((NT + nt) * KSTEPS + ks) * 32 + lane) * 16;
      bfrag[nt] = *(const v16bf*)bp;
    }
#pragma unroll
    for (int mt = 0; mt < 2; ++mt)
#pragma unroll
      for (int nt = 0; nt < 2; ++nt)
        acc[mt][nt] = __builtin_amdgcn_wmma_f32_16x16x32_bf16(
            false, a[mt], false, bfrag[nt], (short)0, acc[mt][nt], false, false);
  }

  // epilogue: sigmoid + transposed store. C layout: lane holds column
  // (lane&15) of rows r + half*8, r = 0..7 -> 8 consecutive batch rows.
#pragma unroll
  for (int mt = 0; mt < 2; ++mt)
#pragma unroll
    for (int nt = 0; nt < 2; ++nt) {
      const int col = (NT + nt) * 16 + mrow;
      const int rb  = Rbase + mt * 16 + half * 8;
      float* dst = pT + (size_t)col * BATCH + rb;
      const v8f c = acc[mt][nt];
      *(float4*)(dst)     = make_float4(sigf(c[0]), sigf(c[1]), sigf(c[2]), sigf(c[3]));
      *(float4*)(dst + 4) = make_float4(sigf(c[4]), sigf(c[5]), sigf(c[6]), sigf(c[7]));
    }
}

// ---------------------------------------------------------------------------
// Kernel 3: tree pass. Each thread owns 2 batch rows. Since
// mu_parent = mu_left + mu_right, only leaf sums are needed for the penalty.
// Leaves grouped by depth-7 ancestor: 7 + 7 coalesced float2 loads -> 8 leaf
// mus. y_pred accumulated in registers (W_leaf cached in LDS); leaf sums
// reduced with shfl_xor butterfly into per-wave LDS rows (deterministic).
// ---------------------------------------------------------------------------
__global__ __launch_bounds__(256)
void sdt_tree(const float* __restrict__ pT,
              const float* __restrict__ Wl,
              float* __restrict__ out,
              float* __restrict__ Spart) {
  __shared__ float sWl[10 * 1024];
  __shared__ float sS[8][1024];
  const int tid  = threadIdx.x;
  const int lane = tid & 31;
  const int wave = tid >> 5;
  for (int i = tid; i < 10 * 1024; i += 256) sWl[i] = Wl[i];
  for (int i = tid; i < 8 * 1024; i += 256) (&sS[0][0])[i] = 0.0f;
  __syncthreads();

  const int b0 = (blockIdx.x * 256 + tid) * 2;   // rows b0, b0+1

  float acc0[10], acc1[10];
#pragma unroll
  for (int o = 0; o < 10; ++o) { acc0[o] = 0.f; acc1[o] = 0.f; }

  for (int g = 0; g < 128; ++g) {
    const int h7 = 128 + g;                      // depth-7 internal node (heap)
    float m7a = 1.f, m7b = 1.f;
#pragma unroll
    for (int d = 0; d < 7; ++d) {
      const int h = h7 >> (7 - d);               // ancestor at depth d (root=1)
      const float2 pv = *(const float2*)(pT + (size_t)(h - 1) * BATCH + b0);
      const int bit = (h7 >> (6 - d)) & 1;       // even child => prob p
      m7a *= bit ? (1.f - pv.x) : pv.x;
      m7b *= bit ? (1.f - pv.y) : pv.y;
    }
    const float2 p7  = *(const float2*)(pT + (size_t)(h7 - 1) * BATCH + b0);
    const float2 p8l = *(const float2*)(pT + (size_t)(2 * h7 - 1) * BATCH + b0);
    const float2 p8r = *(const float2*)(pT + (size_t)(2 * h7) * BATCH + b0);
    float2 p9[4];
#pragma unroll
    for (int j = 0; j < 4; ++j)
      p9[j] = *(const float2*)(pT + (size_t)(4 * h7 + j - 1) * BATCH + b0);

    float m8a[2], m8b[2], m9a[4], m9b[4];
    m8a[0] = m7a * p7.x;         m8b[0] = m7b * p7.y;
    m8a[1] = m7a * (1.f - p7.x); m8b[1] = m7b * (1.f - p7.y);
    m9a[0] = m8a[0] * p8l.x;         m9b[0] = m8b[0] * p8l.y;
    m9a[1] = m8a[0] * (1.f - p8l.x); m9b[1] = m8b[0] * (1.f - p8l.y);
    m9a[2] = m8a[1] * p8r.x;         m9b[2] = m8b[1] * p8r.y;
    m9a[3] = m8a[1] * (1.f - p8r.x); m9b[3] = m8b[1] * (1.f - p8r.y);

#pragma unroll
    for (int j = 0; j < 4; ++j) {
#pragma unroll
      for (int s = 0; s < 2; ++s) {
        const float mua = s ? m9a[j] * (1.f - p9[j].x) : m9a[j] * p9[j].x;
        const float mub = s ? m9b[j] * (1.f - p9[j].y) : m9b[j] * p9[j].y;
        const int leaf = g * 8 + j * 2 + s;
        float t = mua + mub;                      // batch-sum via butterfly
        t += __shfl_xor(t, 1, 32);
        t += __shfl_xor(t, 2, 32);
        t += __shfl_xor(t, 4, 32);
        t += __shfl_xor(t, 8, 32);
        t += __shfl_xor(t, 16, 32);
        if (lane == 0) sS[wave][leaf] += t;
#pragma unroll
        for (int o = 0; o < 10; ++o) {
          const float w = sWl[o * 1024 + leaf];
          acc0[o] += mua * w;
          acc1[o] += mub * w;
        }
      }
    }
  }

#pragma unroll
  for (int o = 0; o < 10; ++o) {
    out[(size_t)b0 * 10 + o]       = acc0[o];
    out[(size_t)(b0 + 1) * 10 + o] = acc1[o];
  }
  __syncthreads();
  for (int i = tid; i < 1024; i += 256) {
    float s = 0.f;
#pragma unroll
    for (int w = 0; w < 8; ++w) s += sS[w][i];
    Spart[(size_t)blockIdx.x * 1024 + i] = s;
  }
}

// ---------------------------------------------------------------------------
// Kernel 4: penalty. Sum 64 per-block leaf partials, build heap of node sums
// upward (S_parent = S_left + S_right), then
// penalty = -0.5 * sum_{h=2..2047} lambda*2^{-(depth-1)} * (log a + log(1-a)),
// a = S[h]/S[h>>1].
// ---------------------------------------------------------------------------
__global__ __launch_bounds__(1024)
void sdt_penalty(const float* __restrict__ Spart, float* __restrict__ outp) {
  __shared__ float heap[2048];
  __shared__ float red[1024];
  const int t = threadIdx.x;

  float s = 0.f;
  for (int blk = 0; blk < 64; ++blk) s += Spart[(size_t)blk * 1024 + t];
  heap[1024 + t] = s;
  __syncthreads();
  for (int n = 512; n >= 1; n >>= 1) {
    if (t < n) heap[n + t] = heap[2 * (n + t)] + heap[2 * (n + t) + 1];
    __syncthreads();
  }

  float accp = 0.f;
  for (int h = 2 + t; h < 2048; h += 1024) {
    const int depth = 31 - __clz(h);                 // 1..10
    const float coeff = 1e-3f * exp2f(-(float)(depth - 1));
    const float alpha = heap[h] / heap[h >> 1];
    accp += coeff * (logf(alpha) + logf(1.0f - alpha));
  }
  red[t] = accp;
  __syncthreads();
  for (int n = 512; n >= 1; n >>= 1) {
    if (t < n) red[t] += red[t + n];
    __syncthreads();
  }
  if (t == 0) outp[0] = -0.5f * red[0];
}

// ---------------------------------------------------------------------------
extern "C" void kernel_launch(void* const* d_in, const int* in_sizes, int n_in,
                              void* d_out, int out_size, void* d_ws, size_t ws_size,
                              hipStream_t stream) {
  (void)in_sizes; (void)n_in; (void)out_size; (void)ws_size;
  const float* X  = (const float*)d_in[0];
  const float* Wi = (const float*)d_in[1];
  const float* Wl = (const float*)d_in[2];
  float* out = (float*)d_out;
  char* ws = (char*)d_ws;

  float*  pT    = (float*)(ws + WS_PT);
  __bf16* Wp    = (__bf16*)(ws + WS_WP);
  float*  bias  = (float*)(ws + WS_BIAS);
  float*  Spart = (float*)(ws + WS_SPART);

  sdt_prepack<<<200, 256, 0, stream>>>(Wi, Wp, bias);
  sdt_gemm_sigmoid<<<dim3(16, 256), 256, 0, stream>>>(X, Wp, bias, pT);
  sdt_tree<<<64, 256, 0, stream>>>(pT, Wl, out, Spart);
  sdt_penalty<<<1, 1024, 0, stream>>>(Spart, out + (size_t)BATCH * 10);
}
